// MultiHeadSelfAttention_62062277427242
// MI455X (gfx1250) — matrix operand order repaired and checked by reading, not compiled, not run on hardware
//
#include <hip/hip_runtime.h>

// Multi-head self-attention, fused flash-style, for MI455X (gfx1250).
// B=2, S=2048, DIM=512, H=8, DH=64. q=k=x@W_in, v=x@W_out, diagonal masked.
// Attention computes S^T = K Q^T and O^T = V^T P^T so each lane owns one
// query column: softmax reductions are in-register + one shfl_xor(16).

#define BATCH 2
#define SEQ   2048
#define DIMSZ 512
#define NH    8
#define DH    64
#define NROW  (BATCH * SEQ)   // 4096

typedef __attribute__((ext_vector_type(16))) __bf16 v16bf;
typedef __attribute__((ext_vector_type(8)))  __bf16 v8bf;
typedef __attribute__((ext_vector_type(8)))  float  v8f;
typedef __attribute__((ext_vector_type(4)))  float  v4f;

__device__ __forceinline__ __bf16 f2bf(float f) {
  unsigned u = __builtin_bit_cast(unsigned, f);
  u += 0x7FFFu + ((u >> 16) & 1u);            // round-to-nearest-even
  unsigned short h = (unsigned short)(u >> 16);
  return __builtin_bit_cast(__bf16, h);
}

__device__ __forceinline__ v16bf cat16(v8bf lo, v8bf hi) {
  return __builtin_shufflevector(lo, hi, 0, 1, 2, 3, 4, 5, 6, 7,
                                         8, 9, 10, 11, 12, 13, 14, 15);
}
__device__ __forceinline__ v8bf ld8(const __bf16* p) {   // 16B-aligned b128 load
  return *(const v8bf*)p;
}

// ---------------------------------------------------------------------------
// fp32 -> bf16 bulk conversion
// ---------------------------------------------------------------------------
__global__ void cvt_f32_bf16(const float* __restrict__ in,
                             __bf16* __restrict__ out, int n) {
  int i = blockIdx.x * blockDim.x + threadIdx.x;
  int stride = gridDim.x * blockDim.x;
  for (; i < n; i += stride) out[i] = f2bf(in[i]);
}

// fp32 [K][N] -> bf16 transposed [N][K] (for K-innermost B-fragment loads)
__global__ void cvt_transpose_w(const float* __restrict__ in,
                                __bf16* __restrict__ out) {
  int idx = blockIdx.x * blockDim.x + threadIdx.x;     // over 512*512
  int k = idx >> 9, n = idx & 511;
  out[(long)n * DIMSZ + k] = f2bf(in[idx]);
}

// ---------------------------------------------------------------------------
// Projection GEMM: Y = X(bf16)[4096x512] @ W(bf16, pre-transposed [N][K])
// blockIdx.z == 0: Q = x @ W_in  -> row-major [4096][512]
// blockIdx.z == 1: V = x @ W_out -> TRANSPOSED Vt [512][4096]
// 4 waves / block, one 16x16 tile per wave, K-loop 512/32 = 16 WMMAs.
// ---------------------------------------------------------------------------
__global__ void __launch_bounds__(128)
proj_gemm(const __bf16* __restrict__ X,
          const __bf16* __restrict__ WinT, const __bf16* __restrict__ WoutT,
          __bf16* __restrict__ Qo, __bf16* __restrict__ Vto) {
  const __bf16* Wt = blockIdx.z ? WoutT : WinT;

  const int wave = threadIdx.x >> 5;
  const int lane = threadIdx.x & 31;
  const int half = lane >> 4;
  const int lidx = lane & 15;
  const int row0 = blockIdx.y * 16;
  const int col0 = blockIdx.x * 64 + wave * 16;

  const __bf16* xrow = X  + (long)(row0 + lidx) * DIMSZ;  // A: M = lane%16
  const __bf16* wrow = Wt + (long)(col0 + lidx) * DIMSZ;  // B: N = lane%16

  v8f acc = {};
  #pragma unroll 1
  for (int kb = 0; kb < DIMSZ; kb += 32) {
    v16bf a = cat16(ld8(xrow + kb + half * 8),
                    ld8(xrow + kb + 16 + half * 8));
    v16bf bm = cat16(ld8(wrow + kb + half * 8),
                     ld8(wrow + kb + 16 + half * 8));
    acc = __builtin_amdgcn_wmma_f32_16x16x32_bf16(false, a, false, bm,
                                                  (short)0, acc, false, false);
  }

  if (blockIdx.z == 0) {
    #pragma unroll
    for (int r = 0; r < 8; ++r) {                        // C: M = r + 8*half
      const int row = row0 + r + half * 8;
      Qo[(long)row * DIMSZ + col0 + lidx] = f2bf(acc[r]);
    }
  } else {
    #pragma unroll
    for (int r = 0; r < 8; ++r) {                        // Vt[d][b*S+s]
      const int row = row0 + r + half * 8;
      Vto[(long)(col0 + lidx) * NROW + row] = f2bf(acc[r]);
    }
  }
}

// ---------------------------------------------------------------------------
// Fused flash attention (transposed-logit formulation).
// One wave handles 16 queries of one (b,h); key tiles of 32 (== WMMA K).
//   S^T[32x16] = K_tile[32x64] @ Q^T[64x16]   (4 WMMAs, keys are M)
//   O^T[64x16] += V^T_tile[64x32] @ P^T[32x16] (4 WMMAs, dims are M)
// Lane owns query column lane%16: per-lane m/l scalars, 2 shuffles/tile.
// Diagonal (q==k) masked with -1e9 per the reference.
// ---------------------------------------------------------------------------
__global__ void __launch_bounds__(32)
flash_attn(const __bf16* __restrict__ Qbf, const __bf16* __restrict__ Vt,
           float* __restrict__ Out) {
  const int lane  = threadIdx.x & 31;
  const int half  = lane >> 4;
  const int lidx  = lane & 15;
  const int qbase = blockIdx.x * 16;
  const int h     = blockIdx.y;
  const int b     = blockIdx.z;
  const long rowOff = (long)b * SEQ;
  const int  hOff   = h * DH;

  __shared__ __bf16 Plds[16 * 32];   // P^T tile: [query][key], 1 KB

  // Q^T B-fragments (N = query = lane%16): d runs half*8+0..7, 16+half*8+0..7 (+32)
  const __bf16* qr = Qbf + (rowOff + qbase + lidx) * DIMSZ + hOff;
  v16bf bq0 = cat16(ld8(qr + half * 8),      ld8(qr + 16 + half * 8));
  v16bf bq1 = cat16(ld8(qr + 32 + half * 8), ld8(qr + 48 + half * 8));

  v8f o0 = {}, o1 = {}, o2 = {}, o3 = {};
  float m_i = -3.0e38f, l_i = 0.0f;   // per-lane (per-query) softmax state
  const float scale = 0.125f;         // 1/sqrt(DH)
  const int q_glob = qbase + lidx;    // this lane's query index

  for (int kk = 0; kk < SEQ; kk += 32) {
    // K-tile A-fragments (M = key): rows kk+lidx and kk+16+lidx
    const __bf16* kr0 = Qbf + (rowOff + kk + lidx) * DIMSZ + hOff;
    const __bf16* kr1 = Qbf + (rowOff + kk + 16 + lidx) * DIMSZ + hOff;

    if (kk + 32 < SEQ) {  // prefetch next key tile (global_prefetch_b8)
      __builtin_prefetch(kr0 + 32 * DIMSZ, 0, 1);
      __builtin_prefetch(kr1 + 32 * DIMSZ, 0, 1);
    }

    v16bf a00 = cat16(ld8(kr0 + half * 8),      ld8(kr0 + 16 + half * 8));
    v16bf a01 = cat16(ld8(kr0 + 32 + half * 8), ld8(kr0 + 48 + half * 8));
    v16bf a10 = cat16(ld8(kr1 + half * 8),      ld8(kr1 + 16 + half * 8));
    v16bf a11 = cat16(ld8(kr1 + 32 + half * 8), ld8(kr1 + 48 + half * 8));

    v8f s0 = {}, s1 = {};   // S^T: s0 keys kk+0..15, s1 keys kk+16..31
    s0 = __builtin_amdgcn_wmma_f32_16x16x32_bf16(false, a00, false, bq0,
                                                 (short)0, s0, false, false);
    s0 = __builtin_amdgcn_wmma_f32_16x16x32_bf16(false, a01, false, bq1,
                                                 (short)0, s0, false, false);
    s1 = __builtin_amdgcn_wmma_f32_16x16x32_bf16(false, a10, false, bq0,
                                                 (short)0, s1, false, false);
    s1 = __builtin_amdgcn_wmma_f32_16x16x32_bf16(false, a11, false, bq1,
                                                 (short)0, s1, false, false);

    // ----- mask + online softmax: lane owns query column q_glob -----
    float v0[8], v1[8];
    float tmax = -3.0e38f;
    #pragma unroll
    for (int r = 0; r < 8; ++r) {
      const int key0 = kk + r + half * 8;        // C: M = r + 8*half
      const int key1 = key0 + 16;
      v0[r] = s0[r] * scale + ((key0 == q_glob) ? -1.0e9f : 0.0f);
      v1[r] = s1[r] * scale + ((key1 == q_glob) ? -1.0e9f : 0.0f);
      tmax = fmaxf(tmax, fmaxf(v0[r], v1[r]));
    }
    tmax = fmaxf(tmax, __shfl_xor(tmax, 16, 32));   // combine key halves
    const float mnew  = fmaxf(m_i, tmax);
    const float alpha = __expf(m_i - mnew);
    m_i = mnew;

    v8bf pk0, pk1;
    float psum = 0.0f;
    #pragma unroll
    for (int r = 0; r < 8; ++r) {
      const float p0 = __expf(v0[r] - mnew);
      const float p1 = __expf(v1[r] - mnew);
      psum += p0 + p1;
      pk0[r] = f2bf(p0);
      pk1[r] = f2bf(p1);
    }
    psum += __shfl_xor(psum, 16, 32);
    l_i = l_i * alpha + psum;

    #pragma unroll
    for (int r = 0; r < 8; ++r) {   // rescale O^T (query column = lane)
      o0[r] *= alpha; o1[r] *= alpha; o2[r] *= alpha; o3[r] *= alpha;
    }

    // stage P^T[query][key]: keys r+8*half and 16+r+8*half -> 2 b128 stores
    *(v8bf*)&Plds[lidx * 32 + 8 * half]      = pk0;
    *(v8bf*)&Plds[lidx * 32 + 16 + 8 * half] = pk1;
    __syncthreads();  // single-wave WG: cheap; orders LDS writes vs reads

    // P^T B-fragment (N = query): keys half*8+0..7, 16+half*8+0..7 -> 2 ds_load_b128
    v16bf bp = cat16(ld8(&Plds[lidx * 32 + half * 8]),
                     ld8(&Plds[lidx * 32 + 16 + half * 8]));
    __syncthreads();

    // ----- O^T += V^T P^T; V^T A-frag from Vt[d][b*S+s] (M = d) -----
    #pragma unroll
    for (int t = 0; t < 4; ++t) {
      const __bf16* vr = Vt + (long)(hOff + t * 16 + lidx) * NROW
                            + rowOff + kk;
      v16bf av = cat16(ld8(vr + half * 8), ld8(vr + 16 + half * 8));
      v8f& ot = (t == 0) ? o0 : (t == 1) ? o1 : (t == 2) ? o2 : o3;
      ot = __builtin_amdgcn_wmma_f32_16x16x32_bf16(false, av, false, bp,
                                                   (short)0, ot, false, false);
    }
  }

  // ----- epilogue: per-lane query row; d contiguous -> b128 f32 stores -----
  const float inv = 1.0f / l_i;
  float* orow = Out + (rowOff + q_glob) * DIMSZ + hOff;
  #pragma unroll
  for (int t = 0; t < 4; ++t) {
    const v8f& ot = (t == 0) ? o0 : (t == 1) ? o1 : (t == 2) ? o2 : o3;
    v4f lo4 = { ot[0] * inv, ot[1] * inv, ot[2] * inv, ot[3] * inv };
    v4f hi4 = { ot[4] * inv, ot[5] * inv, ot[6] * inv, ot[7] * inv };
    *(v4f*)&orow[t * 16 + 8 * half]     = lo4;   // O^T: d = t*16 + 8*half + r
    *(v4f*)&orow[t * 16 + 8 * half + 4] = hi4;
  }
}

// ---------------------------------------------------------------------------
extern "C" void kernel_launch(void* const* d_in, const int* in_sizes, int n_in,
                              void* d_out, int out_size, void* d_ws, size_t ws_size,
                              hipStream_t stream) {
  const float* x     = (const float*)d_in[0];
  const float* W_in  = (const float*)d_in[1];
  const float* W_out = (const float*)d_in[2];
  float* out = (float*)d_out;

  // workspace (bf16): x 4MB | q 4MB | vT 4MB | W_inT 0.5MB | W_outT 0.5MB
  char* ws = (char*)d_ws;
  __bf16* x_bf  = (__bf16*)(ws);
  __bf16* q_bf  = (__bf16*)(ws + (size_t)4  * 1024 * 1024);
  __bf16* vt_bf = (__bf16*)(ws + (size_t)8  * 1024 * 1024);
  __bf16* wi_bf = (__bf16*)(ws + (size_t)12 * 1024 * 1024);
  __bf16* wo_bf = (__bf16*)(ws + (size_t)12 * 1024 * 1024 + 512 * 1024);

  const int NX = BATCH * SEQ * DIMSZ;   // 2,097,152
  const int NW = DIMSZ * DIMSZ;         //   262,144
  cvt_f32_bf16<<<dim3(512), dim3(256), 0, stream>>>(x, x_bf, NX);
  cvt_transpose_w<<<dim3(NW / 256), dim3(256), 0, stream>>>(W_in,  wi_bf);
  cvt_transpose_w<<<dim3(NW / 256), dim3(256), 0, stream>>>(W_out, wo_bf);

  // Q (row-major) and V (transposed) projections; z selects the weight
  proj_gemm<<<dim3(DIMSZ / 64, NROW / 16, 2), dim3(128), 0, stream>>>(
      x_bf, wi_bf, wo_bf, q_bf, vt_bf);

  // Fused masked attention, one wave per 16-query tile per (b,h)
  flash_attn<<<dim3(SEQ / 16, NH, BATCH), dim3(32), 0, stream>>>(
      q_bf, vt_bf, out);
}
